// Detector_41180146434828
// MI455X (gfx1250) — 8-bit matrix operand order repaired and checked by reading, compiled, not run on hardware
//
#include <hip/hip_runtime.h>
#include <stdint.h>

#define NC      80
#define NCAND   10647
#define KSEL    512
#define THRESHF 0.5f
#define IOU_T   0.3f

typedef int   v16i __attribute__((ext_vector_type(16)));
typedef float v8f  __attribute__((ext_vector_type(8)));

__device__ __forceinline__ float sigmoidf(float x) { return 1.0f / (1.0f + expf(-x)); }

__device__ __forceinline__ unsigned int orderKey(float f) {
    unsigned int u = __float_as_uint(f);
    return (u & 0x80000000u) ? ~u : (u | 0x80000000u);
}

__device__ __forceinline__ void mapIdx(int n, int& sc, int& H, int& h, int& w, int& a) {
    if (n < 507)       { sc = 0; H = 13; }
    else if (n < 2535) { sc = 1; H = 26; n -= 507; }
    else               { sc = 2; H = 52; n -= 2535; }
    a = n % 3;
    int hw = n / 3;
    w = hw % H;
    h = hw / H;
}

__global__ __launch_bounds__(1024)
void yolo_select_decode(const float* __restrict__ o13, const float* __restrict__ o26,
                        const float* __restrict__ o52,
                        const float* __restrict__ an13, const float* __restrict__ an26,
                        const float* __restrict__ an52,
                        float* __restrict__ bkW, float* __restrict__ valsW,
                        float* __restrict__ outp)
{
    __shared__ unsigned int       skey[NCAND];
    __shared__ unsigned long long rbest[1024];
    __shared__ int                selIdx[KSEL];

    const int b   = blockIdx.x;
    const int tid = threadIdx.x;

    for (int n = tid; n < NCAND; n += 1024) {
        int sc, H, h, w, a;
        mapIdx(n, sc, H, h, w, a);
        const float* src = (sc == 0) ? o13 : (sc == 1) ? o26 : o52;
        long HW = (long)H * H;
        float logit = src[(long)b * 255 * HW + (long)(a * 85 + 4) * HW + h * H + w];
        float conf  = sigmoidf(logit);
        float score = (conf > THRESHF) ? conf : -1.0f;
        skey[n] = orderKey(score);
    }
    __syncthreads();

    for (int it = 0; it < KSEL; ++it) {
        unsigned long long best = 0ull;
        for (int n = tid; n < NCAND; n += 1024) {
            unsigned long long k =
                ((unsigned long long)skey[n] << 14) | (unsigned int)(16383 - n);
            if (k > best) best = k;
        }
        rbest[tid] = best;
        __syncthreads();
        for (int off = 512; off > 0; off >>= 1) {
            if (tid < off) {
                unsigned long long o = rbest[tid + off];
                if (o > rbest[tid]) rbest[tid] = o;
            }
            __syncthreads();
        }
        if (tid == 0) {
            int n = 16383 - (int)(rbest[0] & 16383u);
            selIdx[it] = n;
            skey[n] = 0u;
        }
        __syncthreads();
    }

    if (tid < KSEL) {
        int n = selIdx[tid];
        int sc, H, h, w, a;
        mapIdx(n, sc, H, h, w, a);
        const float* src; const float* anc; float t;
        if (sc == 0)      { src = o13; anc = an13; t = 32.0f; }
        else if (sc == 1) { src = o26; anc = an26; t = 16.0f; }
        else              { src = o52; anc = an52; t = 8.0f;  }
        long HW   = (long)H * H;
        long base = (long)b * 255 * HW + (long)(a * 85) * HW + h * H + w;

        float o0 = src[base];
        float o1 = src[base + HW];
        float o2 = src[base + 2 * HW];
        float o3 = src[base + 3 * HW];
        float o4 = src[base + 4 * HW];
        float conf = sigmoidf(o4);
        float cx = ((float)w + sigmoidf(o0)) * t;
        float cy = ((float)h + sigmoidf(o1)) * t;
        float bw = anc[a * 2 + 0] * expf(o2);
        float bh = anc[a * 2 + 1] * expf(o3);

        float bestv = -3.0e38f; int bestc = 0;
        for (int c = 0; c < NC; ++c) {
            float v = src[base + (long)(5 + c) * HW];
            if (v > bestv) { bestv = v; bestc = c; }
        }

        float row[6] = { cx - 0.5f * bw, cy - 0.5f * bh,
                         cx + 0.5f * bw, cy + 0.5f * bh,
                         conf, (float)bestc };

        float* bkp = bkW + ((long)b * KSEL + tid) * 6;
        float* op  = outp + ((long)b * KSEL + tid) * 7;
        #pragma unroll
        for (int c = 0; c < 6; ++c) { bkp[c] = row[c]; op[c] = row[c]; }
        valsW[b * KSEL + tid] = conf;
    }
}

__global__ __launch_bounds__(256)
void yolo_suppress(const float* __restrict__ bkW, unsigned char* __restrict__ supW)
{
    __shared__ float sx1[KSEL], sy1[KSEL], sx2[KSEL], sy2[KSEL], sar[KSEL];
    __shared__ int   scl[KSEL];

    const int b   = blockIdx.x;
    const int tid = threadIdx.x;

    for (int i = tid; i < KSEL; i += 256) {
        const float* r = bkW + ((long)b * KSEL + i) * 6;
        float x1 = r[0], y1 = r[1], x2 = r[2], y2 = r[3];
        sx1[i] = x1; sy1[i] = y1; sx2[i] = x2; sy2[i] = y2;
        sar[i] = (x2 - x1) * (y2 - y1);
        scl[i] = (int)r[5];
    }
    __syncthreads();

    const int lane = tid & 31;
    const int wv   = tid >> 5;
    const int half = lane >> 4;
    const int lm   = lane & 15;
    const int tr   = blockIdx.y * 8 + wv;

    const int clsA = scl[tr * 16 + lm];
    v16i A;
    #pragma unroll
    for (int vi = 0; vi < 16; ++vi) {
        int v  = vi & 7;
        int kb = ((vi >> 3) & 1) * 64 + (v & 1) * 4 + ((v >> 1) & 1) * 16
               + ((v >> 2) & 1) * 32 + half * 8;
        int d = clsA - kb;
        A[vi] = (d >= 0 && d < 4) ? (0x38 << (8 * d)) : 0;
    }

    float rx1[8], ry1[8], rx2[8], ry2[8], rar[8];
    #pragma unroll
    for (int r = 0; r < 8; ++r) {
        int gi = tr * 16 + half * 8 + r;
        rx1[r] = sx1[gi]; ry1[r] = sy1[gi];
        rx2[r] = sx2[gi]; ry2[r] = sy2[gi]; rar[r] = sar[gi];
    }

    unsigned char* sup = supW + (size_t)b * KSEL * KSEL;

    for (int tc = 0; tc < 32; ++tc) {
        const int gj   = tc * 16 + lm;
        const int clsB = scl[gj];
        v16i Bf;
        #pragma unroll
        for (int vi = 0; vi < 16; ++vi) {
            int kb = (vi >> 3) * 64 + (vi & 1) * 4 + ((vi >> 1) & 1) * 16 + ((vi >> 2) & 1) * 32 + half * 8;
            int d  = clsB - kb;
            Bf[vi] = (d >= 0 && d < 4) ? (0x38 << (8 * d)) : 0;
        }
        v8f Cacc = {};
        v8f Dv = __builtin_amdgcn_wmma_f32_16x16x128_fp8_fp8(
                     A, Bf, (short)0, Cacc, false, false);

        float cx1 = sx1[gj], cy1 = sy1[gj], cx2 = sx2[gj], cy2 = sy2[gj], car = sar[gj];
        #pragma unroll
        for (int r = 0; r < 8; ++r) {
            int gi = tr * 16 + half * 8 + r;
            float ix1 = fmaxf(rx1[r], cx1);
            float iy1 = fmaxf(ry1[r], cy1);
            float ix2 = fminf(rx2[r], cx2);
            float iy2 = fminf(ry2[r], cy2);
            float inter = fmaxf(ix2 - ix1, 0.0f) * fmaxf(iy2 - iy1, 0.0f);
            float iou = inter / (rar[r] + car - inter + 1e-9f);
            bool s = (Dv[r] > 0.5f) && (iou > IOU_T);
            sup[(size_t)gi * KSEL + gj] = s ? 1 : 0;
        }
    }
}

__global__ __launch_bounds__(512)
void yolo_nms_scan(const float* __restrict__ valsW, const unsigned char* __restrict__ supW,
                   float* __restrict__ outp)
{
    __shared__ unsigned char keepS[KSEL];
    __shared__ unsigned char validS[KSEL];
    __shared__ int flag;

    const int b = blockIdx.x;
    const int j = threadIdx.x;

    keepS[j]  = 0;
    validS[j] = (valsW[b * KSEL + j] > THRESHF) ? 1 : 0;
    const unsigned char* sup = supW + (size_t)b * KSEL * KSEL;
    __syncthreads();

    for (int i = 0; i < KSEL; ++i) {
        if (j == 0) flag = 0;
        __syncthreads();
        if (keepS[j] && sup[(size_t)i * KSEL + j]) flag = 1;
        __syncthreads();
        if (j == 0) keepS[i] = (validS[i] && !flag) ? 1 : 0;
        __syncthreads();
    }
    outp[((size_t)b * KSEL + j) * 7 + 6] = (float)keepS[j];
}

extern "C" void kernel_launch(void* const* d_in, const int* in_sizes, int n_in,
                              void* d_out, int out_size, void* d_ws, size_t ws_size,
                              hipStream_t stream)
{
    (void)in_sizes; (void)n_in; (void)out_size; (void)ws_size;
    const float* o13  = (const float*)d_in[0];
    const float* o26  = (const float*)d_in[1];
    const float* o52  = (const float*)d_in[2];
    const float* an13 = (const float*)d_in[3];
    const float* an26 = (const float*)d_in[4];
    const float* an52 = (const float*)d_in[5];
    float* outp = (float*)d_out;

    char* ws = (char*)d_ws;
    float*         bkW   = (float*)ws;
    float*         valsW = (float*)(ws + (size_t)32 * KSEL * 6 * 4);
    unsigned char* supW  = (unsigned char*)(ws + (size_t)32 * KSEL * 6 * 4
                                               + (size_t)32 * KSEL * 4);

    yolo_select_decode<<<dim3(32), dim3(1024), 0, stream>>>(
        o13, o26, o52, an13, an26, an52, bkW, valsW, outp);
    yolo_suppress<<<dim3(32, 4), dim3(256), 0, stream>>>(bkW, supW);
    yolo_nms_scan<<<dim3(32), dim3(512), 0, stream>>>(valsW, supW, outp);
}
